// BoundaryTransformerLayer_35656818491920
// MI455X (gfx1250) — matrix operand order repaired and checked by reading, not compiled, not run on hardware
//
#include <hip/hip_runtime.h>
#include <hip/hip_bf16.h>
#include <math.h>

typedef __attribute__((ext_vector_type(16))) _Float16 v16h;
typedef __attribute__((ext_vector_type(8)))  float    v8f;

#define CH   64
#define NSAM 16
#define CS   8
#define EPSB 1e-5f

// ---------------------------------------------------------------------------
// Kernel 1: QKV projections via WMMA f32<-f16 16x16x32.
// One wave = one 16-row tile of x; computes all 64 cols for Q, K, V.
// ---------------------------------------------------------------------------
__global__ __launch_bounds__(256)
void qkv_wmma_kernel(const float* __restrict__ x,
                     const float* __restrict__ Wq, const float* __restrict__ bq,
                     const float* __restrict__ Wk, const float* __restrict__ bk,
                     const float* __restrict__ Wv, const float* __restrict__ bv,
                     float* __restrict__ oq, float* __restrict__ ok,
                     float* __restrict__ ov, int n)
{
    const int wave = (blockIdx.x * blockDim.x + threadIdx.x) >> 5;
    const int lane = threadIdx.x & 31;
    const int m0   = wave * 16;
    if (m0 >= n) return;                      // wave-uniform: EXEC stays all-ones

    // ---- A tiles: x rows m0..m0+15, two K-chunks (K 0..31, 32..63), f16 ----
    // ISA 16-bit A 16x32 layout: lanes 0-15 -> M=lane, K {0..7,16..23};
    //                            lanes 16-31 -> M=lane-16, K {8..15,24..31}.
    const int mrow = m0 + (lane & 15);
    const int alo  = (lane < 16) ? 0 : 8;
    v16h a0, a1;
#pragma unroll
    for (int i = 0; i < 16; ++i) {
        const int kl = (i < 8) ? (alo + i) : (16 + alo + (i - 8));
        a0[i] = (_Float16)x[mrow * CH + kl];
        a1[i] = (_Float16)x[mrow * CH + 32 + kl];
    }

    // ---- B layout: lanes 0-15 hold K {0..7,16..23}, lanes 16-31 hold K {8..15,24..31},
    //      N = lane&15, same K order as A ----
    const int ncol = lane & 15;
    const int kb   = (lane < 16) ? 0 : 8;
    const int rbase = m0 + ((lane < 16) ? 0 : 8);

    const float* const Ws[3] = {Wq, Wk, Wv};
    const float* const Bs[3] = {bq, bk, bv};
    float* const       Os[3] = {oq, ok, ov};

#pragma unroll
    for (int w = 0; w < 3; ++w) {
        const float* __restrict__ W  = Ws[w];
        float* __restrict__       O  = Os[w];
        const float               bb = Bs[w][0]; (void)bb;
#pragma unroll
        for (int nt = 0; nt < 4; ++nt) {
            const int n0 = nt * 16;
            v16h b0, b1;
#pragma unroll
            for (int i = 0; i < 16; ++i) {
                const int kbi = (i < 8) ? (kb + i) : (16 + kb + (i - 8));
                b0[i] = (_Float16)W[(kbi) * CH      + n0 + ncol];
                b1[i] = (_Float16)W[(32 + kbi) * CH + n0 + ncol];
            }
            v8f c = {};
            c = __builtin_amdgcn_wmma_f32_16x16x32_f16(false, a0, false, b0,
                                                       (short)0, c, false, false);
            c = __builtin_amdgcn_wmma_f32_16x16x32_f16(false, a1, false, b1,
                                                       (short)0, c, false, false);
            const float bias = Bs[w][n0 + ncol];
#pragma unroll
            for (int r = 0; r < 8; ++r)
                O[(size_t)(rbase + r) * CH + n0 + ncol] = c[r] + bias;
        }
    }
}

// ---------------------------------------------------------------------------
// Kernel 2: fused gather + position MLP + attention-weight MLP + softmax +
// weighted aggregation. Thread = (point, neighbor); 16 lanes per point.
// ---------------------------------------------------------------------------
__global__ __launch_bounds__(256)
void fused_attn_kernel(const float* __restrict__ p,
                       const int*   __restrict__ idx,
                       const float* __restrict__ xq,
                       const float* __restrict__ xk,
                       const float* __restrict__ xv,
                       const float* __restrict__ Wp1, const float* __restrict__ bp1,
                       const float* __restrict__ bnp_g, const float* __restrict__ bnp_b,
                       const float* __restrict__ bnp_m, const float* __restrict__ bnp_v,
                       const float* __restrict__ Wp2, const float* __restrict__ bp2,
                       const float* __restrict__ bn1_g, const float* __restrict__ bn1_b,
                       const float* __restrict__ bn1_m, const float* __restrict__ bn1_v,
                       const float* __restrict__ Ww1, const float* __restrict__ bw1,
                       const float* __restrict__ bn2_g, const float* __restrict__ bn2_b,
                       const float* __restrict__ bn2_m, const float* __restrict__ bn2_v,
                       const float* __restrict__ Ww2, const float* __restrict__ bw2,
                       float* __restrict__ out, int n)
{
    __shared__ float sWp2[3 * CH], sbp2[CH], sS1[CH], sB1[CH];
    __shared__ float sWw1[CH * CS], sbw1[CS], sS2[CS], sB2[CS];
    __shared__ float sWw2[CS * CS], sbw2[CS];
    __shared__ float sWp1[9], sbp1[3], sSp[3], sBp[3];

    const int tid = threadIdx.x;
    for (int i = tid; i < 3 * CH; i += 256) sWp2[i] = Wp2[i];
    for (int i = tid; i < CH * CS; i += 256) sWw1[i] = Ww1[i];
    for (int i = tid; i < CH; i += 256) {
        sbp2[i] = bp2[i];
        const float sc = bn1_g[i] * rsqrtf(bn1_v[i] + EPSB);
        sS1[i] = sc;
        sB1[i] = bn1_b[i] - bn1_m[i] * sc;
    }
    if (tid < CS * CS) sWw2[tid] = Ww2[tid];
    if (tid < CS) {
        sbw1[tid] = bw1[tid];
        sbw2[tid] = bw2[tid];
        const float sc = bn2_g[tid] * rsqrtf(bn2_v[tid] + EPSB);
        sS2[tid] = sc;
        sB2[tid] = bn2_b[tid] - bn2_m[tid] * sc;
    }
    if (tid < 9) sWp1[tid] = Wp1[tid];
    if (tid < 3) {
        sbp1[tid] = bp1[tid];
        const float sc = bnp_g[tid] * rsqrtf(bnp_v[tid] + EPSB);
        sSp[tid] = sc;
        sBp[tid] = bnp_b[tid] - bnp_m[tid] * sc;
    }
    __syncthreads();

    const int pt = blockIdx.x * 16 + (tid >> 4);
    const int s  = tid & 15;
    if (pt >= n) return;

    const int nb = idx[(size_t)pt * NSAM + s];

    // relative position
    const float gx0 = p[(size_t)nb * 3 + 0] - p[(size_t)pt * 3 + 0];
    const float gx1 = p[(size_t)nb * 3 + 1] - p[(size_t)pt * 3 + 1];
    const float gx2 = p[(size_t)nb * 3 + 2] - p[(size_t)pt * 3 + 2];

    // linear_p stage 1: Linear(3,3) -> BN -> ReLU
    float t3[3];
#pragma unroll
    for (int j = 0; j < 3; ++j) {
        float u = gx0 * sWp1[0 * 3 + j] + gx1 * sWp1[1 * 3 + j] +
                  gx2 * sWp1[2 * 3 + j] + sbp1[j];
        u = u * sSp[j] + sBp[j];
        t3[j] = fmaxf(u, 0.0f);
    }

    const float4* kg4 = (const float4*)(xk + (size_t)nb * CH);
    const float4* vg4 = (const float4*)(xv + (size_t)nb * CH);
    const float4* xq4 = (const float4*)(xq + (size_t)pt * CH);

    float aj[CS];
#pragma unroll
    for (int j = 0; j < CS; ++j) aj[j] = sbw1[j];

    float vpr[CH];

#pragma unroll
    for (int c4 = 0; c4 < CH / 4; ++c4) {
        const float4 kg = kg4[c4];
        const float4 vv = vg4[c4];
        const float4 qq = xq4[c4];
        const float kgx[4] = {kg.x, kg.y, kg.z, kg.w};
        const float vvx[4] = {vv.x, vv.y, vv.z, vv.w};
        const float qqx[4] = {qq.x, qq.y, qq.z, qq.w};
#pragma unroll
        for (int e = 0; e < 4; ++e) {
            const int c = c4 * 4 + e;
            // pr = relu-path positional encoding projected to COUT
            const float pr = sbp2[c] + t3[0] * sWp2[0 * CH + c] +
                                       t3[1] * sWp2[1 * CH + c] +
                                       t3[2] * sWp2[2 * CH + c];
            const float w  = kgx[e] - qqx[e] + pr;        // kg - xq + pr
            const float h  = fmaxf(w * sS1[c] + sB1[c], 0.0f);
#pragma unroll
            for (int j = 0; j < CS; ++j) aj[j] = fmaf(h, sWw1[c * CS + j], aj[j]);
            vpr[c] = vvx[e] + pr;                          // vg + pr
        }
    }

    // second attention MLP stage: BN -> ReLU -> Linear(8,8)
    float h2[CS];
#pragma unroll
    for (int j = 0; j < CS; ++j) h2[j] = fmaxf(aj[j] * sS2[j] + sB2[j], 0.0f);
    float w2[CS];
#pragma unroll
    for (int j = 0; j < CS; ++j) {
        float acc = sbw2[j];
#pragma unroll
        for (int i = 0; i < CS; ++i) acc = fmaf(h2[i], sWw2[i * CS + j], acc);
        w2[j] = acc;
    }

    // softmax over the 16 neighbors (16-lane groups within the wave32)
#pragma unroll
    for (int j = 0; j < CS; ++j) {
        float m = w2[j];
#pragma unroll
        for (int d = 1; d < 16; d <<= 1) m = fmaxf(m, __shfl_xor(m, d, 16));
        const float e = __expf(w2[j] - m);
        float sum = e;
#pragma unroll
        for (int d = 1; d < 16; d <<= 1) sum += __shfl_xor(sum, d, 16);
        w2[j] = e / sum;
    }

    // out[pt][c] = sum_s vpr[s][c] * w2[s][c & 7]
#pragma unroll
    for (int c = 0; c < CH; ++c) {
        float v = vpr[c] * w2[c & 7];
#pragma unroll
        for (int d = 1; d < 16; d <<= 1) v += __shfl_xor(v, d, 16);
        if ((c >> 2) == s) out[(size_t)pt * CH + c] = v;   // coalesced: 4 ch/lane
    }
}

// ---------------------------------------------------------------------------
extern "C" void kernel_launch(void* const* d_in, const int* in_sizes, int n_in,
                              void* d_out, int out_size, void* d_ws, size_t ws_size,
                              hipStream_t stream)
{
    const float* p   = (const float*)d_in[0];
    const float* x   = (const float*)d_in[1];
    const int*   idx = (const int*)  d_in[2];
    const float* Wq  = (const float*)d_in[3];  const float* bq = (const float*)d_in[4];
    const float* Wk  = (const float*)d_in[5];  const float* bk = (const float*)d_in[6];
    const float* Wv  = (const float*)d_in[7];  const float* bv = (const float*)d_in[8];
    const float* Wp1 = (const float*)d_in[9];  const float* bp1 = (const float*)d_in[10];
    const float* bnp_g = (const float*)d_in[11]; const float* bnp_b = (const float*)d_in[12];
    const float* bnp_m = (const float*)d_in[13]; const float* bnp_v = (const float*)d_in[14];
    const float* Wp2 = (const float*)d_in[15]; const float* bp2 = (const float*)d_in[16];
    const float* bn1_g = (const float*)d_in[17]; const float* bn1_b = (const float*)d_in[18];
    const float* bn1_m = (const float*)d_in[19]; const float* bn1_v = (const float*)d_in[20];
    const float* Ww1 = (const float*)d_in[21]; const float* bw1 = (const float*)d_in[22];
    const float* bn2_g = (const float*)d_in[23]; const float* bn2_b = (const float*)d_in[24];
    const float* bn2_m = (const float*)d_in[25]; const float* bn2_v = (const float*)d_in[26];
    const float* Ww2 = (const float*)d_in[27]; const float* bw2 = (const float*)d_in[28];

    const int n = in_sizes[1] / CH;            // N = 100000
    float* out = (float*)d_out;

    float* ws = (float*)d_ws;
    float* xq = ws;                            // N*64 each
    float* xk = ws + (size_t)n * CH;
    float* xv = ws + 2 * (size_t)n * CH;

    // Kernel 1: WMMA QKV. One wave per 16-row tile, 8 waves per block.
    const int tiles  = (n + 15) / 16;
    const int blocks1 = (tiles + 7) / 8;
    qkv_wmma_kernel<<<blocks1, 256, 0, stream>>>(x, Wq, bq, Wk, bk, Wv, bv,
                                                 xq, xk, xv, n);

    // Kernel 2: fused gather/MLP/softmax/aggregate. 16 points per 256-thread block.
    const int blocks2 = (n + 15) / 16;
    fused_attn_kernel<<<blocks2, 256, 0, stream>>>(
        p, idx, xq, xk, xv,
        Wp1, bp1, bnp_g, bnp_b, bnp_m, bnp_v, Wp2, bp2,
        bn1_g, bn1_b, bn1_m, bn1_v, Ww1, bw1,
        bn2_g, bn2_b, bn2_m, bn2_v, Ww2, bw2,
        out, n);
}
